// PedTransformer_13030930776160
// MI455X (gfx1250) — matrix operand order repaired and checked by reading, not compiled, not run on hardware
//
#include <hip/hip_runtime.h>
#include <hip/hip_bf16.h>
#include <math.h>

typedef __attribute__((ext_vector_type(16))) _Float16 v16h;
typedef __attribute__((ext_vector_type(8)))  _Float16 v8h;
typedef __attribute__((ext_vector_type(8)))  float    v8f;

#define NCAMS 7
#define EMB   512
#define NQ    1024
#define HF    135
#define WF    240
#define NHEAD 4
#define HD    128
#define ROWS  (NQ * NCAMS)   // 7168

// ---------------------------------------------------------------------------
// Tiled weight convert + transpose: W (K x N, f32) -> Wt (N x K, f16)
// Coalesced reads and writes via a 32x32 LDS tile. K, N multiples of 32.
// grid (N/32, K/32), block (32, 8)
// ---------------------------------------------------------------------------
__global__ void wT_f16_tiled_kernel(const float* __restrict__ W, _Float16* __restrict__ Wt,
                                    int K, int N) {
    __shared__ float tile[32][33];
    const int n0 = blockIdx.x * 32;
    const int k0 = blockIdx.y * 32;
    const int tx = threadIdx.x;
    const int ty = threadIdx.y;
    #pragma unroll
    for (int i = 0; i < 32; i += 8)
        tile[ty + i][tx] = W[(size_t)(k0 + ty + i) * N + n0 + tx];
    __syncthreads();
    #pragma unroll
    for (int i = 0; i < 32; i += 8)
        Wt[(size_t)(n0 + ty + i) * K + k0 + tx] = (_Float16)tile[tx][ty + i];
}

// ---------------------------------------------------------------------------
// Query MLP: h = relu(LN(queries) @ w1 + b1); gc = sigmoid(h@w2+b2)*[480,1440]
// Then per-cam 3x3 projection -> coords (gx,gy) + valid mask.
// One block (128 threads) per query.
// ---------------------------------------------------------------------------
__global__ void qmlp_proj_kernel(const float* __restrict__ queries,
                                 const float* __restrict__ ln0_g,
                                 const float* __restrict__ ln0_b,
                                 const float* __restrict__ w1,
                                 const float* __restrict__ b1,
                                 const float* __restrict__ w2,
                                 const float* __restrict__ b2,
                                 const float* __restrict__ proj,   // (7,3,3)
                                 float* __restrict__ coords,       // (7,NQ,2)
                                 int*   __restrict__ valid) {      // (7,NQ)
    __shared__ float xs[EMB];
    __shared__ float red[128];
    __shared__ float hs[128];
    __shared__ float gcs[2];
    const int q = blockIdx.x;
    const int t = threadIdx.x;

    const float* row = queries + (size_t)q * EMB;
    float s = 0.f;
    for (int i = t; i < EMB; i += 128) { float v = row[i]; xs[i] = v; s += v; }
    red[t] = s; __syncthreads();
    for (int off = 64; off > 0; off >>= 1) { if (t < off) red[t] += red[t + off]; __syncthreads(); }
    const float mean = red[0] / (float)EMB;
    __syncthreads();
    float vv = 0.f;
    for (int i = t; i < EMB; i += 128) { float d = xs[i] - mean; vv += d * d; }
    red[t] = vv; __syncthreads();
    for (int off = 64; off > 0; off >>= 1) { if (t < off) red[t] += red[t + off]; __syncthreads(); }
    const float rstd = rsqrtf(red[0] / (float)EMB + 1e-5f);
    __syncthreads();
    for (int i = t; i < EMB; i += 128) xs[i] = (xs[i] - mean) * rstd * ln0_g[i] + ln0_b[i];
    __syncthreads();

    // h[t] = relu(sum_e xs[e] * w1[e,t] + b1[t])
    float acc = b1[t];
    for (int e = 0; e < EMB; ++e) acc += xs[e] * w1[e * 128 + t];
    hs[t] = fmaxf(acc, 0.f);
    __syncthreads();

    if (t < 2) {
        float a = b2[t];
        for (int j = 0; j < 128; ++j) a += hs[j] * w2[j * 2 + t];
        float sg = 1.f / (1.f + expf(-a));
        gcs[t] = sg * (t == 0 ? 480.f : 1440.f);
    }
    __syncthreads();

    if (t < NCAMS) {
        const float* P = proj + t * 9;
        const float gx0 = gcs[0], gy0 = gcs[1];
        float xx = P[0] * gx0 + P[1] * gy0 + P[2];
        float yy = P[3] * gx0 + P[4] * gy0 + P[5];
        float zz = P[6] * gx0 + P[7] * gy0 + P[8];
        int vld = (zz > 1e-5f) ? 1 : 0;
        float zc = fmaxf(zz, 1e-5f);
        float gx = (xx / zc / 1920.f - 0.5f) * 2.f;
        float gy = (yy / zc / 1080.f - 0.5f) * 2.f;
        vld = vld && (gx > -1.f) && (gx < 1.f) && (gy > -1.f) && (gy < 1.f);
        coords[((size_t)t * NQ + q) * 2 + 0] = gx;
        coords[((size_t)t * NQ + q) * 2 + 1] = gy;
        valid[(size_t)t * NQ + q] = vld;
    }
}

// ---------------------------------------------------------------------------
// Bilinear sampling: write f16 directly into the scrambled x layout
//   Xh[e*7168 + q*7 + c]  (== row-major (7168 x 512) X after the ref reshape)
// grid (NQ, NCAMS), block 256 over channels
// ---------------------------------------------------------------------------
__global__ void sample_kernel(const float* __restrict__ feats,   // (7,512,135,240)
                              const float* __restrict__ coords,  // (7,NQ,2)
                              _Float16* __restrict__ Xh) {
    const int q = blockIdx.x;
    const int c = blockIdx.y;
    const float gx = coords[((size_t)c * NQ + q) * 2 + 0];
    const float gy = coords[((size_t)c * NQ + q) * 2 + 1];
    const float ix = ((gx + 1.f) * (float)WF - 1.f) * 0.5f;
    const float iy = ((gy + 1.f) * (float)HF - 1.f) * 0.5f;
    const float x0f = floorf(ix), y0f = floorf(iy);
    const float wx = ix - x0f, wy = iy - y0f;
    const float x1f = x0f + 1.f, y1f = y0f + 1.f;

    const float in00 = (x0f >= 0.f && x0f < (float)WF && y0f >= 0.f && y0f < (float)HF) ? 1.f : 0.f;
    const float in10 = (x1f >= 0.f && x1f < (float)WF && y0f >= 0.f && y0f < (float)HF) ? 1.f : 0.f;
    const float in01 = (x0f >= 0.f && x0f < (float)WF && y1f >= 0.f && y1f < (float)HF) ? 1.f : 0.f;
    const float in11 = (x1f >= 0.f && x1f < (float)WF && y1f >= 0.f && y1f < (float)HF) ? 1.f : 0.f;
    const int x0 = (int)fminf(fmaxf(x0f, 0.f), (float)(WF - 1));
    const int x1 = (int)fminf(fmaxf(x1f, 0.f), (float)(WF - 1));
    const int y0 = (int)fminf(fmaxf(y0f, 0.f), (float)(HF - 1));
    const int y1 = (int)fminf(fmaxf(y1f, 0.f), (float)(HF - 1));
    const float w00 = (1.f - wx) * (1.f - wy) * in00;
    const float w10 = wx * (1.f - wy) * in10;
    const float w01 = (1.f - wx) * wy * in01;
    const float w11 = wx * wy * in11;

    const float* fb = feats + (size_t)c * EMB * HF * WF;
    for (int e = threadIdx.x; e < EMB; e += blockDim.x) {
        const float* fc = fb + (size_t)e * HF * WF;
        float v = w00 * fc[y0 * WF + x0] + w10 * fc[y0 * WF + x1]
                + w01 * fc[y1 * WF + x0] + w11 * fc[y1 * WF + x1];
        if (isnan(v)) v = 0.f;   // nan_to_num
        Xh[(size_t)e * ROWS + (size_t)q * NCAMS + c] = (_Float16)v;
    }
}

// ---------------------------------------------------------------------------
// WMMA GEMM, register-blocked: each wave computes a 64x32 macro-tile
// (4x2 grid of 16x16 WMMA tiles), ~21 flops/byte of operand traffic.
//   C(MxN,f32) = A(MxK,f16 row-major) @ Bt(NxK,f16 row-major)^T
//   + bias[N] (opt) + residual(MxN f32, opt); optional f16 copy of C.
// One wave (32 threads) per block. grid = (N/32, M/64). K % 32 == 0.
// ---------------------------------------------------------------------------
__global__ void gemm_wmma_kernel(const _Float16* __restrict__ A,
                                 const _Float16* __restrict__ Bt,
                                 const float* __restrict__ bias,
                                 const float* __restrict__ residual,
                                 float* __restrict__ C,
                                 _Float16* __restrict__ Ch,
                                 int K, int N) {
    const int lane = threadIdx.x;            // 0..31
    const int l15  = lane & 15;
    const int hi   = lane >> 4;              // 0 or 1
    const int m0   = blockIdx.y * 64;
    const int n0   = blockIdx.x * 32;
    const int hiK  = hi * 16;                // A half-K offset for high lanes

    const _Float16* arow0 = A  + (size_t)(m0 + l15) * K;
    const _Float16* brow0 = Bt + (size_t)(n0 + l15) * K;

    union AF { v16h v; v8h h[2]; };
    v8f acc[4][2] = {};

    for (int k0 = 0; k0 < K; k0 += 32) {
        AF a[4];
        #pragma unroll
        for (int i = 0; i < 4; ++i) {
            const _Float16* ar = arow0 + (size_t)(i * 16) * K;
            a[i].h[0] = *(const v8h*)(ar + k0 + hiK);        // halves 0..7  : K = k0+hiK+0..7
            a[i].h[1] = *(const v8h*)(ar + k0 + 8 + hiK);    // halves 8..15 : K = k0+8+hiK+0..7
        }
        v16h b[2];
        #pragma unroll
        for (int j = 0; j < 2; ++j)
            b[j] = *(const v16h*)(brow0 + (size_t)(j * 16) * K + k0 + hi * 16);
        #pragma unroll
        for (int i = 0; i < 4; ++i)
            #pragma unroll
            for (int j = 0; j < 2; ++j)
                acc[i][j] = __builtin_amdgcn_wmma_f32_16x16x32_f16(
                    false, a[i].v, false, b[j], (short)0, acc[i][j], false, false);
    }

    #pragma unroll
    for (int j = 0; j < 2; ++j) {
        const int cn = n0 + j * 16 + l15;
        const float bv = bias ? bias[cn] : 0.f;
        #pragma unroll
        for (int i = 0; i < 4; ++i) {
            #pragma unroll
            for (int r = 0; r < 8; ++r) {
                const int cm = m0 + i * 16 + r + hi * 8;
                float v = acc[i][j][r] + bv;
                if (residual) v += residual[(size_t)cm * N + cn];
                C[(size_t)cm * N + cn] = v;
                if (Ch) Ch[(size_t)cm * N + cn] = (_Float16)v;
            }
        }
    }
}

// ---------------------------------------------------------------------------
// Attention: per (Q, head). qkv f32 (7168 x 1536). Output Oh f16 (7168 x 512).
// ---------------------------------------------------------------------------
__global__ void attention_kernel(const float* __restrict__ QKV,
                                 const int* __restrict__ valid,   // (7,NQ)
                                 _Float16* __restrict__ Oh) {
    __shared__ float qs[NCAMS][HD];
    __shared__ float ks[NCAMS][HD];
    __shared__ float vs[NCAMS][HD];
    __shared__ float sc[NCAMS][NCAMS];
    __shared__ float att[NCAMS][NCAMS];
    const int Q = blockIdx.x;
    const int h = blockIdx.y;
    const int d = threadIdx.x;               // 128 threads

    for (int n = 0; n < NCAMS; ++n) {
        const float* base = QKV + (size_t)(Q * NCAMS + n) * (3 * EMB);
        qs[n][d] = base[h * HD + d];
        ks[n][d] = base[EMB + h * HD + d];
        vs[n][d] = base[2 * EMB + h * HD + d];
    }
    __syncthreads();

    if (d < NCAMS * NCAMS) {
        const int n = d / NCAMS, m = d % NCAMS;
        float s = 0.f;
        for (int e = 0; e < HD; ++e) s += qs[n][e] * ks[m][e];
        s *= rsqrtf((float)HD);
        if (!valid[(size_t)m * NQ + Q]) s = -INFINITY;
        sc[n][m] = s;
    }
    __syncthreads();

    if (d < NCAMS) {
        float mx = -INFINITY;
        for (int m = 0; m < NCAMS; ++m) mx = fmaxf(mx, sc[d][m]);
        float sum = 0.f;
        float ex[NCAMS];
        for (int m = 0; m < NCAMS; ++m) { ex[m] = expf(sc[d][m] - mx); sum += ex[m]; }
        for (int m = 0; m < NCAMS; ++m) att[d][m] = ex[m] / sum;   // all-masked -> NaN (matches ref)
    }
    __syncthreads();

    for (int n = 0; n < NCAMS; ++n) {
        float a = 0.f;
        for (int m = 0; m < NCAMS; ++m) a += att[n][m] * vs[m][d];
        Oh[(size_t)(Q * NCAMS + n) * EMB + h * HD + d] = (_Float16)a;
    }
}

// ---------------------------------------------------------------------------
// Mean over cameras: Meanh[q,e] = mean_c AttnOut[(q*7+c), e]  (f16 out)
// ---------------------------------------------------------------------------
__global__ void mean_kernel(const float* __restrict__ AttnOut, _Float16* __restrict__ Meanh) {
    const int q = blockIdx.x;
    for (int e = threadIdx.x; e < EMB; e += blockDim.x) {
        float s = 0.f;
        for (int c = 0; c < NCAMS; ++c) s += AttnOut[(size_t)(q * NCAMS + c) * EMB + e];
        Meanh[(size_t)q * EMB + e] = (_Float16)(s * (1.f / (float)NCAMS));
    }
}

// ---------------------------------------------------------------------------
// LayerNorm + ReLU over rows of 512, in place. One block (256) per row.
// ---------------------------------------------------------------------------
__global__ void lnrelu_kernel(float* __restrict__ X,
                              const float* __restrict__ g,
                              const float* __restrict__ b) {
    __shared__ float xs[EMB];
    __shared__ float red[256];
    const int row = blockIdx.x;
    const int t = threadIdx.x;
    float* rp = X + (size_t)row * EMB;
    float s = 0.f;
    for (int i = t; i < EMB; i += 256) { float v = rp[i]; xs[i] = v; s += v; }
    red[t] = s; __syncthreads();
    for (int off = 128; off > 0; off >>= 1) { if (t < off) red[t] += red[t + off]; __syncthreads(); }
    const float mean = red[0] / (float)EMB;
    __syncthreads();
    float vv = 0.f;
    for (int i = t; i < EMB; i += 256) { float d = xs[i] - mean; vv += d * d; }
    red[t] = vv; __syncthreads();
    for (int off = 128; off > 0; off >>= 1) { if (t < off) red[t] += red[t + off]; __syncthreads(); }
    const float rstd = rsqrtf(red[0] / (float)EMB + 1e-5f);
    __syncthreads();
    for (int i = t; i < EMB; i += 256)
        rp[i] = fmaxf((xs[i] - mean) * rstd * g[i] + b[i], 0.f);
}

// ---------------------------------------------------------------------------
// Final heads: logits = c @ cls_w2 + cls_b2 ; boxes = sigmoid(r @ reg_w2 + reg_b2)
// d_out: [0..2047] logits, [2048..4095] boxes. One block (128) per query.
// ---------------------------------------------------------------------------
__global__ void head_kernel(const float* __restrict__ C, const float* __restrict__ R,
                            const float* __restrict__ cw2, const float* __restrict__ cb2,
                            const float* __restrict__ rw2, const float* __restrict__ rb2,
                            float* __restrict__ out) {
    __shared__ float s0[128], s1[128], s2[128], s3[128];
    const int q = blockIdx.x;
    const int t = threadIdx.x;
    float a0 = 0.f, a1 = 0.f, a2 = 0.f, a3 = 0.f;
    for (int e = t; e < EMB; e += 128) {
        const float cv = C[(size_t)q * EMB + e];
        const float rv = R[(size_t)q * EMB + e];
        a0 += cv * cw2[e * 2 + 0];
        a1 += cv * cw2[e * 2 + 1];
        a2 += rv * rw2[e * 2 + 0];
        a3 += rv * rw2[e * 2 + 1];
    }
    s0[t] = a0; s1[t] = a1; s2[t] = a2; s3[t] = a3; __syncthreads();
    for (int off = 64; off > 0; off >>= 1) {
        if (t < off) { s0[t] += s0[t+off]; s1[t] += s1[t+off]; s2[t] += s2[t+off]; s3[t] += s3[t+off]; }
        __syncthreads();
    }
    if (t == 0) {
        out[q * 2 + 0] = s0[0] + cb2[0];
        out[q * 2 + 1] = s1[0] + cb2[1];
        float b0 = s2[0] + rb2[0], b1 = s3[0] + rb2[1];
        out[2 * NQ + q * 2 + 0] = 1.f / (1.f + expf(-b0));
        out[2 * NQ + q * 2 + 1] = 1.f / (1.f + expf(-b1));
    }
}

// ---------------------------------------------------------------------------
extern "C" void kernel_launch(void* const* d_in, const int* in_sizes, int n_in,
                              void* d_out, int out_size, void* d_ws, size_t ws_size,
                              hipStream_t stream) {
    const float* img_feats = (const float*)d_in[0];
    const float* proj_mat  = (const float*)d_in[1];   // (1,7,3,3)
    const float* queries   = (const float*)d_in[2];
    const float* ln0_g     = (const float*)d_in[3];
    const float* ln0_b     = (const float*)d_in[4];
    const float* w1        = (const float*)d_in[5];
    const float* b1        = (const float*)d_in[6];
    const float* w2        = (const float*)d_in[7];
    const float* b2        = (const float*)d_in[8];
    const float* in_w      = (const float*)d_in[9];
    const float* in_b      = (const float*)d_in[10];
    const float* out_w     = (const float*)d_in[11];
    const float* out_b     = (const float*)d_in[12];
    const float* proj_w    = (const float*)d_in[13];
    const float* proj_b    = (const float*)d_in[14];
    const float* cls_w1    = (const float*)d_in[15];
    const float* cls_b1    = (const float*)d_in[16];
    const float* cls_g     = (const float*)d_in[17];
    const float* cls_beta  = (const float*)d_in[18];
    const float* cls_w2    = (const float*)d_in[19];
    const float* cls_b2    = (const float*)d_in[20];
    const float* reg_w1    = (const float*)d_in[21];
    const float* reg_b1    = (const float*)d_in[22];
    const float* reg_g     = (const float*)d_in[23];
    const float* reg_beta  = (const float*)d_in[24];
    const float* reg_w2    = (const float*)d_in[25];
    const float* reg_b2    = (const float*)d_in[26];
    float* out = (float*)d_out;

    // ---- workspace layout (256B aligned slices) ----
    char* p = (char*)d_ws;
    auto alloc = [&](size_t bytes) -> void* {
        void* r = (void*)p;
        p += (bytes + 255) & ~(size_t)255;
        return r;
    };
    _Float16* Xh     = (_Float16*)alloc((size_t)ROWS * EMB * 2);      // scrambled x, f16
    float*    QKV    = (float*)   alloc((size_t)ROWS * 3 * EMB * 4);
    _Float16* Oh     = (_Float16*)alloc((size_t)ROWS * EMB * 2);      // attention out, f16
    float*    AttnO  = (float*)   alloc((size_t)ROWS * EMB * 4);      // after out_w
    _Float16* Meanh  = (_Float16*)alloc((size_t)NQ * EMB * 2);
    float*    Ofin   = (float*)   alloc((size_t)NQ * EMB * 4);        // o (f32)
    _Float16* Ofh    = (_Float16*)alloc((size_t)NQ * EMB * 2);        // o (f16)
    float*    Cpre   = (float*)   alloc((size_t)NQ * EMB * 4);
    float*    Rpre   = (float*)   alloc((size_t)NQ * EMB * 4);
    float*    coords = (float*)   alloc((size_t)NCAMS * NQ * 2 * 4);
    int*      valid  = (int*)     alloc((size_t)NCAMS * NQ * 4);
    _Float16* inWT   = (_Float16*)alloc((size_t)EMB * 3 * EMB * 2);   // (1536 x 512)
    _Float16* outWT  = (_Float16*)alloc((size_t)EMB * EMB * 2);
    _Float16* projWT = (_Float16*)alloc((size_t)EMB * EMB * 2);
    _Float16* clsWT  = (_Float16*)alloc((size_t)EMB * EMB * 2);
    _Float16* regWT  = (_Float16*)alloc((size_t)EMB * EMB * 2);

    // ---- 1) weight convert+transpose to f16 (tiled, coalesced) ----
    {
        dim3 blk(32, 8);
        wT_f16_tiled_kernel<<<dim3((3 * EMB) / 32, EMB / 32), blk, 0, stream>>>(in_w, inWT, EMB, 3 * EMB);
        dim3 g2(EMB / 32, EMB / 32);
        wT_f16_tiled_kernel<<<g2, blk, 0, stream>>>(out_w,  outWT,  EMB, EMB);
        wT_f16_tiled_kernel<<<g2, blk, 0, stream>>>(proj_w, projWT, EMB, EMB);
        wT_f16_tiled_kernel<<<g2, blk, 0, stream>>>(cls_w1, clsWT,  EMB, EMB);
        wT_f16_tiled_kernel<<<g2, blk, 0, stream>>>(reg_w1, regWT,  EMB, EMB);
    }

    // ---- 2) query MLP + projection -> coords/valid ----
    qmlp_proj_kernel<<<NQ, 128, 0, stream>>>(queries, ln0_g, ln0_b, w1, b1, w2, b2,
                                             proj_mat, coords, valid);

    // ---- 3) bilinear sampling into scrambled f16 X ----
    sample_kernel<<<dim3(NQ, NCAMS), 256, 0, stream>>>(img_feats, coords, Xh);

    // ---- 4) QKV = X @ in_w + in_b  (7168 x 1536) ----
    gemm_wmma_kernel<<<dim3((3 * EMB) / 32, ROWS / 64), 32, 0, stream>>>(
        Xh, inWT, in_b, nullptr, QKV, nullptr, EMB, 3 * EMB);

    // ---- 5) masked multi-head attention ----
    attention_kernel<<<dim3(NQ, NHEAD), HD, 0, stream>>>(QKV, valid, Oh);

    // ---- 6) AttnO = O @ out_w + out_b  (7168 x 512) ----
    gemm_wmma_kernel<<<dim3(EMB / 32, ROWS / 64), 32, 0, stream>>>(
        Oh, outWT, out_b, nullptr, AttnO, nullptr, EMB, EMB);

    // ---- 7) mean over cameras ----
    mean_kernel<<<NQ, 256, 0, stream>>>(AttnO, Meanh);

    // ---- 8) o = Mean @ proj_w + proj_b + queries  (1024 x 512, f32 + f16) ----
    gemm_wmma_kernel<<<dim3(EMB / 32, NQ / 64), 32, 0, stream>>>(
        Meanh, projWT, proj_b, queries, Ofin, Ofh, EMB, EMB);

    // ---- 9) cls / reg pre-activations ----
    gemm_wmma_kernel<<<dim3(EMB / 32, NQ / 64), 32, 0, stream>>>(
        Ofh, clsWT, cls_b1, nullptr, Cpre, nullptr, EMB, EMB);
    gemm_wmma_kernel<<<dim3(EMB / 32, NQ / 64), 32, 0, stream>>>(
        Ofh, regWT, reg_b1, nullptr, Rpre, nullptr, EMB, EMB);

    // ---- 10) LN + ReLU (in place) ----
    lnrelu_kernel<<<NQ, 256, 0, stream>>>(Cpre, cls_g, cls_beta);
    lnrelu_kernel<<<NQ, 256, 0, stream>>>(Rpre, reg_g, reg_beta);

    // ---- 11) heads -> d_out (logits then boxes) ----
    head_kernel<<<NQ, 128, 0, stream>>>(Cpre, Rpre, cls_w2, cls_b2, reg_w2, reg_b2, out);
}
